// VectorQuantizer_55972013802225
// MI455X (gfx1250) — matrix operand order repaired and checked by reading, not compiled, not run on hardware
//
#include <hip/hip_runtime.h>

typedef __attribute__((ext_vector_type(16))) __bf16 v16bf;
typedef __attribute__((ext_vector_type(8)))  float  v8f;

union BFrag { v16bf v; unsigned int u[8]; uint4 q[2]; };
union Acc8  { v8f v; float f[8]; };

__device__ __forceinline__ unsigned int pack_bf16x2(float a, float b) {
  // round-to-nearest-even f32 -> bf16, packed low|high
  unsigned int ua = __builtin_bit_cast(unsigned int, a);
  unsigned int ub = __builtin_bit_cast(unsigned int, b);
  ua = (ua + 0x7FFFu + ((ua >> 16) & 1u)) >> 16;
  ub = (ub + 0x7FFFu + ((ub >> 16) & 1u)) >> 16;
  return (ua & 0xFFFFu) | (ub << 16);
}

// ---- Kernel 0: codebook -> bf16 copy, fp32 norms, zero loss accumulator ----
__global__ __launch_bounds__(256)
void cb_prep_kernel(const float* __restrict__ cb, unsigned short* __restrict__ cbbf,
                    float* __restrict__ cbnorm, float* __restrict__ lossAcc) {
  const int k = blockIdx.x;
  const int t = threadIdx.x;
  float v = cb[(size_t)k * 256 + t];
  unsigned int u = __builtin_bit_cast(unsigned int, v);
  u = (u + 0x7FFFu + ((u >> 16) & 1u)) >> 16;
  cbbf[(size_t)k * 256 + t] = (unsigned short)u;
  float s = v * v;
  #pragma unroll
  for (int off = 16; off >= 1; off >>= 1) s += __shfl_xor(s, off, 32);
  __shared__ float wsum[8];
  if ((t & 31) == 0) wsum[t >> 5] = s;
  __syncthreads();
  if (t == 0) {
    float tot = 0.0f;
    #pragma unroll
    for (int w = 0; w < 8; ++w) tot += wsum[w];
    cbnorm[k] = tot;
    if (k == 0) *lossAcc = 0.0f;
  }
}

// ---- Kernel 1: WMMA distance GEMM + running argmin, async double-buffered --
#define CB_PITCH 264            // bf16 elements per LDS row (pad kills bank stride)
#define CHUNK_ROWS 32           // codebook entries per LDS buffer (2 n-tiles)

// Issue one chunk (32 rows x 512B) of bf16 codebook as async global->LDS b128
// copies; per-lane LDS dest address re-applies the pitch-264 padding.
__device__ __forceinline__ void issue_chunk_async(const unsigned short* cbbf,
                                                  int chunk, unsigned ldsBase,
                                                  int tid) {
  const unsigned long long gbase =
      (unsigned long long)(uintptr_t)cbbf + (unsigned long long)chunk * (CHUNK_ROWS * 512);
  #pragma unroll
  for (int i = 0; i < 4; ++i) {                 // 1024 b128 xfers / 256 threads
    int flat = i * 256 + tid;
    int row = flat >> 5;                        // 0..31
    int seg = flat & 31;                        // 16B segment within row
    unsigned voff  = (unsigned)(row * 512 + seg * 16);
    unsigned laddr = ldsBase + (unsigned)(row * (CB_PITCH * 2) + seg * 16);
    asm volatile("global_load_async_to_lds_b128 %0, %1, %2"
                 :: "v"(laddr), "v"(voff), "s"(gbase) : "memory");
  }
}

__global__ __launch_bounds__(256)
void vq_argmin_kernel(const float* __restrict__ z,
                      const unsigned short* __restrict__ cbbf,
                      const float* __restrict__ cbnorm, int* __restrict__ idxOut) {
  __shared__ unsigned short cbTile[2][CHUNK_ROWS * CB_PITCH];   // ~33 KB

  const int tid  = threadIdx.x;
  const int lane = tid & 31;
  const int wave = tid >> 5;
  const int half = lane >> 4;   // 0: lanes 0-15, 1: lanes 16-31
  const int l15  = lane & 15;

  // Each wave owns a 16-row M tile; 8 waves -> 128 rows per block.
  const int mbase = blockIdx.x * 128 + wave * 16;
  const int m = mbase + l15;
  const float* zr = z + (size_t)(m >> 10) * 262144 + (m & 1023);

  // Build all 8 K-chunk A fragments (bf16 16x32 layout) once, keep in VGPRs.
  BFrag a[8];
  #pragma unroll
  for (int kk = 0; kk < 8; ++kk) {
    #pragma unroll
    for (int v = 0; v < 8; ++v) {
      int c = kk * 32 + ((v & 4) << 2) + half * 8 + 2 * (v & 3);
      float f0 = zr[(size_t)c * 1024];
      float f1 = zr[(size_t)(c + 1) * 1024];
      a[kk].u[v] = pack_bf16x2(f0, f1);
    }
  }

  float best[8];
  int   bidx[8];
  #pragma unroll
  for (int v = 0; v < 8; ++v) { best[v] = 3.4e38f; bidx[v] = 0; }

  const unsigned base0 = (unsigned)(uintptr_t)&cbTile[0][0];
  const unsigned base1 = (unsigned)(uintptr_t)&cbTile[1][0];

  issue_chunk_async(cbbf, 0, base0, tid);       // prologue: chunk 0 -> buf 0

  for (int c = 0; c < 32; ++c) {                // 32 chunks x 32 entries
    if (c < 31) {
      issue_chunk_async(cbbf, c + 1, (c & 1) ? base0 : base1, tid);
      asm volatile("s_wait_asynccnt 0x4" ::: "memory");   // chunk c landed
    } else {
      asm volatile("s_wait_asynccnt 0x0" ::: "memory");
    }
    __syncthreads();                            // all waves' copies visible

    const unsigned short* tile = (c & 1) ? cbTile[1] : cbTile[0];
    #pragma unroll
    for (int nt = 0; nt < 2; ++nt) {
      const int nloc = (nt << 4) + l15;
      // Batch all B fragments first so DS latency is covered by one burst.
      BFrag b[8];
      #pragma unroll
      for (int kk = 0; kk < 8; ++kk) {
        const unsigned short* bp = &tile[nloc * CB_PITCH + kk * 32 + half * 8];
        b[kk].q[0] = *(const uint4*)(bp);
        b[kk].q[1] = *(const uint4*)(bp + 16);
      }
      Acc8 acc;
      #pragma unroll
      for (int v = 0; v < 8; ++v) acc.f[v] = 0.0f;
      #pragma unroll
      for (int kk = 0; kk < 8; ++kk)
        acc.v = __builtin_amdgcn_wmma_f32_16x16x32_bf16(
            false, a[kk].v, false, b[kk].v, (short)0, acc.v, false, false);

      const int n = (c << 5) + (nt << 4) + l15;
      const float cbn = cbnorm[n];
      // D layout: lane holds (M = v + 8*half, N = l15); distance sans |z|^2.
      #pragma unroll
      for (int v = 0; v < 8; ++v) {
        float cand = cbn - 2.0f * acc.f[v];
        if (cand < best[v]) { best[v] = cand; bidx[v] = n; }
      }
    }
    __syncthreads();                            // done reading before overwrite
  }

  // Reduce over the 16 lanes of each half (N residue classes) per M row.
  #pragma unroll
  for (int off = 8; off >= 1; off >>= 1) {
    #pragma unroll
    for (int v = 0; v < 8; ++v) {
      float ob = __shfl_xor(best[v], off, 32);
      int   oi = __shfl_xor(bidx[v], off, 32);
      if (ob < best[v] || (ob == best[v] && oi < bidx[v])) { best[v] = ob; bidx[v] = oi; }
    }
  }
  if (l15 == 0) {
    #pragma unroll
    for (int v = 0; v < 8; ++v)
      idxOut[mbase + half * 8 + v] = bidx[v];
  }
}

// ---------------- Kernel 2: gather z_q (BCHW) + loss partial ----------------
__global__ __launch_bounds__(256)
void vq_gather_loss_kernel(const float* __restrict__ z, const float* __restrict__ cb,
                           const int* __restrict__ idx, float* __restrict__ out,
                           float* __restrict__ lossAcc) {
  __shared__ float rows[32 * 257];
  __shared__ int lidx[32];
  const int tid = threadIdx.x;
  const int mstart = blockIdx.x * 32;
  const int b   = mstart >> 10;
  const int hw0 = mstart & 1023;

  if (tid < 32) lidx[tid] = idx[mstart + tid];
  __syncthreads();
  #pragma unroll 4
  for (int j = 0; j < 32; ++j)                     // coalesced codebook row copy
    rows[j * 257 + tid] = cb[(size_t)lidx[j] * 256 + tid];
  __syncthreads();

  const int cSub = tid >> 5;
  const int hwl  = tid & 31;
  float accv = 0.0f;
  #pragma unroll 4
  for (int c0 = 0; c0 < 32; ++c0) {
    int c = c0 * 8 + cSub;
    size_t gi = (size_t)b * 262144 + (size_t)c * 1024 + (size_t)(hw0 + hwl);
    float v  = rows[hwl * 257 + c];                // conflict-free transpose read
    float zv = z[gi];
    out[gi] = v;                                   // 128B-coalesced stores
    float d = v - zv;
    accv += d * d;
  }
  #pragma unroll
  for (int off = 16; off >= 1; off >>= 1) accv += __shfl_xor(accv, off, 32);
  __shared__ float wsum[8];
  if ((tid & 31) == 0) wsum[tid >> 5] = accv;
  __syncthreads();
  if (tid == 0) {
    float s = 0.0f;
    #pragma unroll
    for (int w = 0; w < 8; ++w) s += wsum[w];
    atomicAdd(lossAcc, s);
  }
}

// ---------------- Kernel 3: finalize scalar loss ----------------------------
__global__ void vq_finalize_kernel(const float* __restrict__ lossAcc,
                                   float* __restrict__ lossOut) {
  *lossOut = 1.25f * (*lossAcc) / 8388608.0f;      // (1+BETA) * mean
}

extern "C" void kernel_launch(void* const* d_in, const int* in_sizes, int n_in,
                              void* d_out, int out_size, void* d_ws, size_t ws_size,
                              hipStream_t stream) {
  const float* z  = (const float*)d_in[0];   // [32,256,32,32]
  const float* cb = (const float*)d_in[1];   // [1024,256]
  float* out = (float*)d_out;                // z_q (8388608) + loss (1)

  unsigned short* cbbf = (unsigned short*)d_ws;                    // 512 KB
  float* cbnorm  = (float*)((char*)d_ws + 524288);                 // 4 KB
  int*   idxbuf  = (int*)((char*)d_ws + 524288 + 4096);            // 128 KB
  float* lossAcc = (float*)((char*)d_ws + 524288 + 4096 + 131072); // 4 B

  cb_prep_kernel<<<1024, 256, 0, stream>>>(cb, cbbf, cbnorm, lossAcc);
  vq_argmin_kernel<<<256, 256, 0, stream>>>(z, cbbf, cbnorm, idxbuf);
  vq_gather_loss_kernel<<<1024, 256, 0, stream>>>(z, cb, idxbuf, out, lossAcc);
  vq_finalize_kernel<<<1, 1, 0, stream>>>(lossAcc, out + 8388608);
}
